// EdgeConv_15101105013037
// MI455X (gfx1250) — matrix operand order repaired and checked by reading, not compiled, not run on hardware
//
#include <hip/hip_runtime.h>

// ---------------------------------------------------------------------------
// EdgeConv fused kernel for MI455X (gfx1250, wave32, WMMA).
//   x'_i = sum_{(i,j) in E} W2^T relu(W1^T [x_i || x_j - x_i] + b1) + b2
// TM=64 edges per workgroup. Gather -> f16 feat tile in LDS -> WMMA GEMM1
// (+bias+ReLU) -> f16 h tile in LDS -> WMMA GEMM2 (+bias) -> fp32 global
// atomic scatter-add.
// Weights are pre-converted to f16 AND packed in WMMA B-fragment order
// (one 1KB contiguous block per (ntile, kblock) covering all 32 lanes):
// each wave's B load is a fully-coalesced contiguous burst, addressed as
// a single per-lane base pointer + immediate offsets (no per-block VALU
// address math -> no WMMA->VALU hazard v_nops in the hot loop).
// ---------------------------------------------------------------------------

typedef __attribute__((ext_vector_type(16))) _Float16 v16h;
typedef __attribute__((ext_vector_type(8)))  _Float16 v8h;
typedef __attribute__((ext_vector_type(8)))  float    v8f;

#define WMMA_F16(A, B, C) \
  __builtin_amdgcn_wmma_f32_16x16x32_f16(false, (A), false, (B), (short)0, (C), false, false)

constexpr int F    = 256;    // node feature dim
constexpr int K1   = 512;    // 2F = GEMM1 K
constexpr int HID  = 512;    // hidden dim
constexpr int OUT  = 256;    // output dim
constexpr int TM   = 64;     // edges per workgroup (4 WMMA M-tiles)
constexpr int LDW  = K1 + 8; // padded LDS row stride (halves): 1040B, 16B aligned

// ---------------------------------------------------------------------------
// Weight packing: fp32 row-major W[k][n] -> f16 fragment-major.
//   block = ntile * (K/32) + kblk
//   WP[(block*32 + L)*16 + t] = W[(kblk*32 + (L>>4)*8 + (t&7) + (t>>3)*16)][ntile*16 + (L&15)]
// Per the ISA B-layout for 16-bit 32x16: lane L holds column (L&15), with
// K rows k0+8h+[0..7] then k0+16+8h+[0..7] (h = L>>4), matching the A fragment.
// ---------------------------------------------------------------------------
__global__ void pack_weights_kernel(const float* __restrict__ W1,  // [K1][HID]
                                    const float* __restrict__ W2,  // [HID][OUT]
                                    _Float16* __restrict__ W1P,
                                    _Float16* __restrict__ W2P) {
  int idx = blockIdx.x * blockDim.x + threadIdx.x;
  if (idx < K1 * HID) {                       // 262144 fragment elements
    const int t     = idx & 15;
    const int L     = (idx >> 4) & 31;
    const int block = idx >> 9;
    const int kblk  = block & 15;             // K1/32 = 16 k-blocks
    const int ntile = block >> 4;             // 32 n-tiles
    const int k = kblk * 32 + (L >> 4) * 8 + (t & 7) + (t >> 3) * 16;
    const int n = ntile * 16 + (L & 15);
    W1P[idx] = (_Float16)W1[k * HID + n];
  }
  if (idx < HID * OUT) {                      // 131072 fragment elements
    const int t     = idx & 15;
    const int L     = (idx >> 4) & 31;
    const int block = idx >> 9;
    const int kblk  = block & 15;             // HID/32 = 16 k-blocks
    const int ntile = block >> 4;             // 16 n-tiles
    const int k = kblk * 32 + (L >> 4) * 8 + (t & 7) + (t >> 3) * 16;
    const int n = ntile * 16 + (L & 15);
    W2P[idx] = (_Float16)W2[k * OUT + n];
  }
}

// A-fragment (16x32 f16): elements 0..7 from p[0..7], 8..15 from p[16..23].
__device__ __forceinline__ v16h load_a_frag(const _Float16* p) {
  v8h lo = *(const v8h*)(p);
  v8h hi = *(const v8h*)(p + 16);
  return __builtin_shufflevector(lo, hi, 0, 1, 2, 3, 4, 5, 6, 7,
                                         8, 9, 10, 11, 12, 13, 14, 15);
}

__global__ __launch_bounds__(256, 1) void edgeconv_kernel(
    const float*    __restrict__ x,     // [N, F] fp32
    const int*      __restrict__ ei,    // [2, E] (row ids, then col ids)
    const float*    __restrict__ b1,    // [HID]
    const float*    __restrict__ b2,    // [OUT]
    const _Float16* __restrict__ W1P,   // packed fragments, 512 KB
    const _Float16* __restrict__ W2P,   // packed fragments, 256 KB
    float*          __restrict__ out,   // [N, OUT] fp32 (zeroed)
    int n_edges)
{
  __shared__ _Float16 sfeat[TM * LDW];  // [64][520] f16  (~66.6 KB)
  __shared__ _Float16 shid [TM * LDW];  // [64][520] f16  (~66.6 KB)
  __shared__ int      srow [TM];

  const int tid  = threadIdx.x;         // 256 threads = 8 wave32
  const int wave = tid >> 5;
  const int lane = tid & 31;
  const int lrow = lane & 15;           // WMMA row/col within half-wave
  const int hsel = lane >> 4;           // 0: lanes 0-15, 1: lanes 16-31
  const int ka   = hsel * 8;            // A-frag K offset per half-wave
  const long long ebase = (long long)blockIdx.x * TM;

  // ---- Phase 1: gather + build feat = [x_i || x_j - x_i] as f16 in LDS ----
  for (int e = 0; e < TM; ++e) {
    long long eidx = ebase + e;
    if (eidx >= n_edges) eidx = n_edges - 1;   // clamp (unused rows masked later)
    const int r = ei[eidx];                    // center node i
    const int c = ei[(long long)n_edges + eidx];
    const float xi = x[(long long)r * F + tid];     // tid covers all 256 cols
    const float xj = x[(long long)c * F + tid];
    sfeat[e * LDW + tid]     = (_Float16)xi;
    sfeat[e * LDW + F + tid] = (_Float16)(xj - xi);
  }
  if (tid < TM) {
    long long eidx = ebase + tid;
    srow[tid] = (eidx < (long long)n_edges) ? ei[eidx] : -1;
  }
  __syncthreads();

  // ---- Phase 2: GEMM1  h = relu(feat @ W1 + b1),  f32 accum, f16 out ------
  // Each wave: all 4 M-tiles x 4 N-tiles (cols wave*64 .. wave*64+63).
  {
    v8f acc[4][4];
#pragma unroll
    for (int m = 0; m < 4; ++m)
#pragma unroll
      for (int n = 0; n < 4; ++n) acc[m][n] = (v8f)0.0f;

    const int nt0 = wave * 4;                  // first n-tile of this wave
    // Per-lane base pointers; all loop offsets are linear in k0 so they fold
    // into the 24-bit instruction immediate (no per-block address VALU).
    const _Float16* abase = sfeat + lrow * LDW + ka;
    const _Float16* bbase = W1P + ((size_t)nt0 * 16 * 32 + lane) * 16;

    for (int k0 = 0; k0 < K1; k0 += 32) {
      v16h a0 = load_a_frag(abase + 0 * 16 * LDW + k0);
      v16h a1 = load_a_frag(abase + 1 * 16 * LDW + k0);
      v16h a2 = load_a_frag(abase + 2 * 16 * LDW + k0);
      v16h a3 = load_a_frag(abase + 3 * 16 * LDW + k0);
#pragma unroll
      for (int n = 0; n < 4; ++n) {
        // halves: kblk*512 == k0*16 ; ntile stride = 16 blocks = 8192 halves
        v16h b = *(const v16h*)(bbase + k0 * 16 + n * 8192);
        acc[0][n] = WMMA_F16(a0, b, acc[0][n]);
        acc[1][n] = WMMA_F16(a1, b, acc[1][n]);
        acc[2][n] = WMMA_F16(a2, b, acc[2][n]);
        acc[3][n] = WMMA_F16(a3, b, acc[3][n]);
      }
    }
    // epilogue: bias + ReLU, write f16 h tile
#pragma unroll
    for (int n = 0; n < 4; ++n) {
      const int col  = (nt0 + n) * 16 + lrow;
      const float bv = b1[col];
#pragma unroll
      for (int m = 0; m < 4; ++m) {
#pragma unroll
        for (int v = 0; v < 8; ++v) {
          float val = acc[m][n][v] + bv;
          val = val > 0.0f ? val : 0.0f;
          shid[(m * 16 + v + hsel * 8) * LDW + col] = (_Float16)val;
        }
      }
    }
  }
  __syncthreads();

  // ---- Phase 3: GEMM2  o = h @ W2 + b2, then atomic scatter-add -----------
  // Each wave: all 4 M-tiles x 2 N-tiles (cols wave*32 .. wave*32+31).
  {
    v8f acc[4][2];
#pragma unroll
    for (int m = 0; m < 4; ++m)
#pragma unroll
      for (int n = 0; n < 2; ++n) acc[m][n] = (v8f)0.0f;

    const int nt0 = wave * 2;                  // first n-tile of this wave
    const _Float16* abase = shid + lrow * LDW + ka;
    const _Float16* bbase = W2P + ((size_t)nt0 * 16 * 32 + lane) * 16;

    for (int k0 = 0; k0 < HID; k0 += 32) {
      v16h a0 = load_a_frag(abase + 0 * 16 * LDW + k0);
      v16h a1 = load_a_frag(abase + 1 * 16 * LDW + k0);
      v16h a2 = load_a_frag(abase + 2 * 16 * LDW + k0);
      v16h a3 = load_a_frag(abase + 3 * 16 * LDW + k0);
#pragma unroll
      for (int n = 0; n < 2; ++n) {
        v16h b = *(const v16h*)(bbase + k0 * 16 + n * 8192);
        acc[0][n] = WMMA_F16(a0, b, acc[0][n]);
        acc[1][n] = WMMA_F16(a1, b, acc[1][n]);
        acc[2][n] = WMMA_F16(a2, b, acc[2][n]);
        acc[3][n] = WMMA_F16(a3, b, acc[3][n]);
      }
    }
    // epilogue: bias + scatter-add into out[row_i]
#pragma unroll
    for (int n = 0; n < 2; ++n) {
      const int col  = (nt0 + n) * 16 + lrow;
      const float bv = b2[col];
#pragma unroll
      for (int m = 0; m < 4; ++m) {
#pragma unroll
        for (int v = 0; v < 8; ++v) {
          const int mrow = m * 16 + v + hsel * 8;
          const int node = srow[mrow];
          if (node >= 0) {
            atomicAdd(out + (size_t)node * OUT + col, acc[m][n][v] + bv);
          }
        }
      }
    }
  }
}

// ---------------------------------------------------------------------------
extern "C" void kernel_launch(void* const* d_in, const int* in_sizes, int n_in,
                              void* d_out, int out_size, void* d_ws, size_t ws_size,
                              hipStream_t stream) {
  const float* x  = (const float*)d_in[0];
  const int*   ei = (const int*)d_in[1];     // edge_index [2, E]
  const float* W1 = (const float*)d_in[2];
  const float* b1 = (const float*)d_in[3];
  const float* W2 = (const float*)d_in[4];
  const float* b2 = (const float*)d_in[5];
  float* out = (float*)d_out;

  const int n_edges = in_sizes[1] / 2;

  _Float16* W1P = (_Float16*)d_ws;                                   // 512 KB
  _Float16* W2P = (_Float16*)((char*)d_ws + (size_t)K1 * HID * 2);   // 256 KB

  // out is accumulated with atomics: zero it (capture-safe async memset).
  hipMemsetAsync(d_out, 0, (size_t)out_size * sizeof(float), stream);

  // Convert + pack weights into WMMA fragment order (262144 threads max).
  pack_weights_kernel<<<(K1 * HID + 255) / 256, 256, 0, stream>>>(W1, W2, W1P, W2P);

  const int n_tiles = (n_edges + TM - 1) / TM;   // 12500 for E=800000
  edgeconv_kernel<<<n_tiles, 256, 0, stream>>>(x, ei, b1, b2, W1P, W2P, out, n_edges);
}
